// OursAttention_51015621542530
// MI455X (gfx1250) — compile-verified, matrix operand order checked by reading, not run on hardware
//
#include <hip/hip_runtime.h>
#include <hip/hip_bf16.h>
#include <float.h>

// ---------------------------------------------------------------------------
// Types for gfx1250 WMMA (wave32).
// ---------------------------------------------------------------------------
typedef __bf16 bf16;
typedef __attribute__((ext_vector_type(16))) __bf16 v16bf;
typedef __attribute__((ext_vector_type(8)))  __bf16 v8bf;
typedef __attribute__((ext_vector_type(8)))  float  v8f;

#define AS3 __attribute__((address_space(3)))

#define NHEAD   16
#define HD      64
#define TTOK    577
#define BATCH   64
#define CDIM    1024
#define C3      3072
#define KKEEP   289          // T - min(288,(T-1)//2)
#define KPAD    320          // KKEEP padded to a multiple of 32 (10 K-chunks)
#define QTILES  19           // ceil(289/16)
#define KTILES  20           // KPAD/16
#define KCHUNKS 10           // KPAD/32
#define MTOK    (BATCH * TTOK)        // 36928 rows (x / qkv)
#define MRED    (BATCH * KKEEP)       // 18496 rows (reduced y)

#define ATTN_WAVES 4                  // q-tiles per attention block
#define ATTN_QGRP  5                  // ceil(QTILES/ATTN_WAVES)
// dynamic LDS partition for attn_kernel
#define LDS_KB_BYTES (KPAD * HD * 2)                         // 40960
#define LDS_SB_BYTES (16 * KPAD * 4)                         // 20480 / wave
#define LDS_PB_BYTES (16 * KPAD * 2)                         // 10240 / wave
#define LDS_ATTN_TOTAL (LDS_KB_BYTES + ATTN_WAVES * (LDS_SB_BYTES + LDS_PB_BYTES)) // 163840

__device__ __forceinline__ v8f wmma_bf16(v16bf a, v16bf b, v8f c) {
  return __builtin_amdgcn_wmma_f32_16x16x32_bf16(
      /*neg_a=*/false, a, /*neg_b=*/false, b, /*c_mod=*/(short)0, c,
      /*reuse_a=*/false, /*reuse_b=*/false);
}

// A/B fragment for V_WMMA_F32_16X16X32_BF16 from a row-major [rows x ld] bf16
// matrix (global or LDS): lane (l&15) selects row, (l>>4) selects K-half:
//   elems 0..7  -> K = k0 + half*8 + {0..7}
//   elems 8..15 -> K = k0 + 16 + half*8 + {0..7}
__device__ __forceinline__ v16bf load_frag_rows(const bf16* p, long row0,
                                                int ld, int k0, int lane) {
  int r = lane & 15;
  int half = lane >> 4;
  const bf16* base = p + (row0 + r) * (long)ld + k0 + half * 8;
  v8bf lo = *(const v8bf*)(base);
  v8bf hi = *(const v8bf*)(base + 16);
  v16bf f;
#pragma unroll
  for (int i = 0; i < 8; ++i) { f[i] = lo[i]; f[8 + i] = hi[i]; }
  return f;
}

// LDS byte offset of a generic pointer into shared memory (for async DMA).
__device__ __forceinline__ unsigned lds_off(void* p) {
  return (unsigned)(unsigned long long)(AS3 char*)p;
}

// CDNA5 async global->LDS copy, 16B per lane, tracked by ASYNCcnt.
__device__ __forceinline__ void async_copy_b128(unsigned lds_byte_off,
                                                const void* gaddr) {
  asm volatile("global_load_async_to_lds_b128 %0, %1, off"
               :
               : "v"(lds_byte_off), "v"((unsigned long long)gaddr)
               : "memory");
}

__device__ __forceinline__ void wait_asynccnt0() {
  asm volatile("s_wait_asynccnt 0x0" ::: "memory");
}

// ---------------------------------------------------------------------------
// f32 -> bf16 convert
// ---------------------------------------------------------------------------
__global__ void cvt_f32_bf16(const float* __restrict__ s, bf16* __restrict__ d,
                             long n) {
  long i = (long)blockIdx.x * blockDim.x + threadIdx.x;
  if (i < n) d[i] = (bf16)s[i];
}

// ---------------------------------------------------------------------------
// C[MxN] = A[MxK] * B[NxK]^T + bias  (bf16 in, f32 accum/out).
// One wave computes a 32x64 output block (2 row-tiles x 4 col-tiles):
// 8 WMMAs per 6 fragment loads. Consecutive waves share the same 64-col B
// panel (M varies fastest) so B fragments stay cache-resident.
// ---------------------------------------------------------------------------
__global__ void gemm_bf16_f32(const bf16* __restrict__ A,
                              const bf16* __restrict__ Bw,
                              const float* __restrict__ bias,
                              float* __restrict__ C, int M, int N, int K) {
  int wave = blockIdx.x * (blockDim.x >> 5) + (threadIdx.x >> 5);
  int lane = threadIdx.x & 31;
  int mblocks = M >> 5;             // 32-row groups
  int nblocks = N >> 6;             // 64-col groups
  if (wave >= mblocks * nblocks) return;  // wave-uniform: EXEC all-ones
  long row0 = (long)(wave % mblocks) * 32;
  long col0 = (long)(wave / mblocks) * 64;

  v8f acc[2][4] = {};
  for (int k0 = 0; k0 < K; k0 += 32) {
    v16bf a0 = load_frag_rows(A, row0, K, k0, lane);
    v16bf a1 = load_frag_rows(A, row0 + 16, K, k0, lane);
#pragma unroll
    for (int t = 0; t < 4; ++t) {
      v16bf b = load_frag_rows(Bw, col0 + 16 * t, K, k0, lane);
      acc[0][t] = wmma_bf16(a0, b, acc[0][t]);
      acc[1][t] = wmma_bf16(a1, b, acc[1][t]);
    }
  }
  int n = lane & 15;
  int rb = (lane >> 4) * 8;
#pragma unroll
  for (int h = 0; h < 2; ++h) {
#pragma unroll
    for (int t = 0; t < 4; ++t) {
      float bs = bias ? bias[col0 + 16 * t + n] : 0.f;
#pragma unroll
      for (int r = 0; r < 8; ++r)
        C[(row0 + 16 * h + rb + r) * (long)N + col0 + 16 * t + n] =
            acc[h][t][r] + bs;
    }
  }
}

// ---------------------------------------------------------------------------
// phi[b,t,h] = ||k_h|| / max(||k_all||, eps)   (unit over heads)
// qkv row layout: [q(1024) | k(1024) | v(1024)]
// ---------------------------------------------------------------------------
__global__ void phi_kernel(const float* __restrict__ qkv,
                           float* __restrict__ phi) {
  int i = blockIdx.x * blockDim.x + threadIdx.x;  // token = b*T + t
  if (i >= MTOK) return;
  const float* krow = qkv + (long)i * C3 + CDIM;
  float mag[NHEAD];
  float ss = 0.f;
#pragma unroll
  for (int h = 0; h < NHEAD; ++h) {
    float s = 0.f;
    for (int d = 0; d < HD; ++d) { float v = krow[h * HD + d]; s += v * v; }
    mag[h] = sqrtf(s);
    ss += s;
  }
  float inv = 1.f / fmaxf(sqrtf(ss), 1e-12f);
#pragma unroll
  for (int h = 0; h < NHEAD; ++h) phi[(long)i * NHEAD + h] = mag[h] * inv;
}

// ---------------------------------------------------------------------------
// Per-batch top-K by max_h phi (CLS pinned), bitonic sort of 1024 in LDS.
// ---------------------------------------------------------------------------
__global__ void topk_kernel(const float* __restrict__ phi,
                            int* __restrict__ keep_abs) {
  __shared__ float s[1024];
  __shared__ int   id[1024];
  int b = blockIdx.x;
  for (int t = threadIdx.x; t < 1024; t += blockDim.x) {
    float sc = -FLT_MAX;
    if (t < TTOK) {
      const float* pr = phi + ((long)b * TTOK + t) * NHEAD;
      sc = pr[0];
      for (int h = 1; h < NHEAD; ++h) sc = fmaxf(sc, pr[h]);
      if (t == 0) sc = FLT_MAX;  // protect CLS
    }
    s[t] = sc;
    id[t] = t;
  }
  __syncthreads();
  for (int k2 = 2; k2 <= 1024; k2 <<= 1) {
    for (int j = k2 >> 1; j > 0; j >>= 1) {
      for (int i = threadIdx.x; i < 1024; i += blockDim.x) {
        int ix = i ^ j;
        if (ix > i) {
          float si = s[i], sx = s[ix];
          int ii = id[i], iv = id[ix];
          bool keep_i = (si > sx) || (si == sx && ii < iv);  // descending
          bool up = ((i & k2) == 0);
          if (up ? !keep_i : keep_i) {
            s[i] = sx; s[ix] = si; id[i] = iv; id[ix] = ii;
          }
        }
      }
      __syncthreads();
    }
  }
  for (int t = threadIdx.x; t < KKEEP; t += blockDim.x)
    keep_abs[b * KKEEP + t] = id[t];
}

// ---------------------------------------------------------------------------
// slot_of[b,t] = argmax_slot phi[t] . phi[keep[slot]]
// ---------------------------------------------------------------------------
__global__ void assign_kernel(const float* __restrict__ phi,
                              const int* __restrict__ keep_abs,
                              int* __restrict__ slot_of) {
  __shared__ float pk[KKEEP][NHEAD];
  int b = blockIdx.x;
  for (int i = threadIdx.x; i < KKEEP * NHEAD; i += blockDim.x) {
    int slot = i / NHEAD, h = i % NHEAD;
    pk[slot][h] =
        phi[((long)b * TTOK + keep_abs[b * KKEEP + slot]) * NHEAD + h];
  }
  __syncthreads();
  for (int t = threadIdx.x; t < TTOK; t += blockDim.x) {
    const float* pt = phi + ((long)b * TTOK + t) * NHEAD;
    float p[NHEAD];
#pragma unroll
    for (int h = 0; h < NHEAD; ++h) p[h] = pt[h];
    float best = -FLT_MAX;
    int bsl = 0;
    for (int slot = 0; slot < KKEEP; ++slot) {
      float d = 0.f;
#pragma unroll
      for (int h = 0; h < NHEAD; ++h) d += p[h] * pk[slot][h];
      if (d > best) { best = d; bsl = slot; }
    }
    slot_of[b * TTOK + t] = bsl;
  }
}

// ---------------------------------------------------------------------------
// Segment-sum merge (exact f32 atomics). One block per token.
// qm/km/vm layout: [B, H, KPAD, HD]
// ---------------------------------------------------------------------------
__global__ void merge_accum(const float* __restrict__ qkv,
                            const int* __restrict__ slot_of,
                            float* __restrict__ qm, float* __restrict__ km,
                            float* __restrict__ vm, float* __restrict__ cnt) {
  int tok = blockIdx.x;
  int b = tok / TTOK;
  int slot = slot_of[tok];
  const float* row = qkv + (long)tok * C3;
  for (int c = threadIdx.x; c < CDIM; c += blockDim.x) {
    int h = c >> 6, d = c & 63;
    long di = (((long)b * NHEAD + h) * KPAD + slot) * HD + d;
    atomicAdd(&qm[di], row[c]);
    atomicAdd(&km[di], row[CDIM + c]);
    atomicAdd(&vm[di], row[2 * CDIM + c]);
  }
  if (threadIdx.x == 0) atomicAdd(&cnt[b * KKEEP + slot], 1.f);
}

// normalize sums -> means; fold softmax scale into q; emit bf16 q/k row-major
// [B,H,KPAD,HD] and V TRANSPOSED [B,H,HD,KPAD] so P@V B-fragments are
// contiguous b128 loads. Padding slots become zeros.
__global__ void merge_norm(const float* __restrict__ qm,
                           const float* __restrict__ km,
                           const float* __restrict__ vm,
                           const float* __restrict__ cnt,
                           bf16* __restrict__ qmb, bf16* __restrict__ kmb,
                           bf16* __restrict__ vmt) {
  long i = (long)blockIdx.x * blockDim.x + threadIdx.x;
  long total = (long)BATCH * NHEAD * KPAD * HD;
  if (i >= total) return;
  long d = i % HD;
  long slot = (i / HD) % KPAD;
  long bh = i / ((long)KPAD * HD);
  long bidx = bh / NHEAD;
  float c = (slot < KKEEP) ? cnt[bidx * KKEEP + slot] : 0.f;
  float inv = (c > 0.f) ? 1.f / c : 0.f;
  const float scale = 0.125f;  // Hd^-0.5
  qmb[i] = (bf16)(qm[i] * inv * scale);
  kmb[i] = (bf16)(km[i] * inv);
  vmt[(bh * HD + d) * KPAD + slot] = (bf16)(vm[i] * inv);
}

// ---------------------------------------------------------------------------
// Attention on the reduced set. Block = 4 waves = 4 q-tiles of one (b,h).
// The block async-DMAs the shared 40KB K panel into LDS (ASYNCcnt), then each
// wave: S = q@k^T (WMMA, K frags from LDS) -> LDS f32; row softmax; P(bf16)
// @ V^T (contiguous global frags) -> y. 160KB dynamic LDS = half a WGP.
// ---------------------------------------------------------------------------
__global__ void attn_kernel(const bf16* __restrict__ qmb,
                            const bf16* __restrict__ kmb,
                            const bf16* __restrict__ vmt,
                            bf16* __restrict__ ymb) {
  extern __shared__ char smem[];
  bf16*  Kb = (bf16*)smem;
  int wid  = threadIdx.x >> 5;
  int lane = threadIdx.x & 31;
  float* Sb = (float*)(smem + LDS_KB_BYTES + wid * LDS_SB_BYTES);
  bf16*  Pb = (bf16*)(smem + LDS_KB_BYTES + ATTN_WAVES * LDS_SB_BYTES +
                      wid * LDS_PB_BYTES);

  int qg = blockIdx.x % ATTN_QGRP;
  int bh = blockIdx.x / ATTN_QGRP;
  int qt = qg * ATTN_WAVES + wid;
  if (qt >= QTILES) qt = QTILES - 1;  // spare wave duplicates tile 18 (benign)

  const bf16* qbase = qmb + (long)bh * KPAD * HD;
  const bf16* kbase = kmb + (long)bh * KPAD * HD;
  const bf16* vbase = vmt + (long)bh * HD * KPAD;

  // cooperative async K-panel stage: KPAD*HD bf16 = 2560 x 16B transfers
  unsigned kb0 = lds_off(Kb);
  for (int it = threadIdx.x; it < (KPAD * HD) / 8; it += ATTN_WAVES * 32)
    async_copy_b128(kb0 + it * 16, kbase + it * 8);
  wait_asynccnt0();
  __syncthreads();

  // q-tile fragments (Hd=64 -> two 32-wide K chunks)
  v16bf qa0 = load_frag_rows(qbase, qt * 16, HD, 0, lane);
  v16bf qa1 = load_frag_rows(qbase, qt * 16, HD, 32, lane);

  int n = lane & 15;
  int rb = (lane >> 4) * 8;
  for (int kt = 0; kt < KTILES; ++kt) {
    v16bf kb_0 = load_frag_rows(Kb, kt * 16, HD, 0, lane);
    v16bf kb_1 = load_frag_rows(Kb, kt * 16, HD, 32, lane);
    v8f acc = {};
    acc = wmma_bf16(qa0, kb_0, acc);
    acc = wmma_bf16(qa1, kb_1, acc);
#pragma unroll
    for (int r = 0; r < 8; ++r) Sb[(rb + r) * KPAD + kt * 16 + n] = acc[r];
  }
  __threadfence_block();  // intra-wave LDS ordering (per-wave private Sb)

  if (lane < 16) {  // one lane per q-row (re-converges before WMMA below)
    float m = -FLT_MAX;
    for (int c = 0; c < KKEEP; ++c) m = fmaxf(m, Sb[lane * KPAD + c]);
    float sum = 0.f;
    for (int c = 0; c < KKEEP; ++c) {
      float e = __expf(Sb[lane * KPAD + c] - m);
      sum += e;
      Sb[lane * KPAD + c] = e;
    }
    float inv = 1.f / sum;
    for (int c = 0; c < KKEEP; ++c)
      Pb[lane * KPAD + c] = (bf16)(Sb[lane * KPAD + c] * inv);
    for (int c = KKEEP; c < KPAD; ++c) Pb[lane * KPAD + c] = (bf16)0.f;
  }
  __threadfence_block();

  v8f yacc[4] = {};
  for (int kc = 0; kc < KCHUNKS; ++kc) {
    v16bf pa = load_frag_rows(Pb, 0, KPAD, kc * 32, lane);
#pragma unroll
    for (int nt = 0; nt < 4; ++nt) {
      // V^T rows are head-dims, contiguous over keys
      v16bf vb = load_frag_rows(vbase, nt * 16, KPAD, kc * 32, lane);
      yacc[nt] = wmma_bf16(pa, vb, yacc[nt]);
    }
  }

  int b = bh / NHEAD, h = bh % NHEAD;
#pragma unroll
  for (int nt = 0; nt < 4; ++nt) {
#pragma unroll
    for (int r = 0; r < 8; ++r) {
      int qrow = qt * 16 + rb + r;
      if (qrow < KKEEP) {
        long orow = (long)b * KKEEP + qrow;
        ymb[orow * CDIM + h * HD + nt * 16 + n] = (bf16)yacc[nt][r];
      }
    }
  }
}

// ---------------------------------------------------------------------------
// Unmerge: out[b,t,:] = yproj[b, slot_of[b,t], :]  (float4 gathers)
// ---------------------------------------------------------------------------
__global__ void unmerge_kernel(const float* __restrict__ yproj,
                               const int* __restrict__ slot_of,
                               float* __restrict__ out) {
  long i = (long)blockIdx.x * blockDim.x + threadIdx.x;
  long total = (long)MTOK * (CDIM / 4);
  if (i >= total) return;
  long c4 = i % (CDIM / 4);
  long tok = i / (CDIM / 4);
  int b = (int)(tok / TTOK);
  int slot = slot_of[tok];
  const float4* src =
      (const float4*)(yproj + ((long)b * KKEEP + slot) * CDIM) + c4;
  ((float4*)out)[i] = *src;
}

// ---------------------------------------------------------------------------
// Host-side launch. Workspace (~756 MB): the 454 MB f32 QKV scratch is dead
// after merge_accum, so all post-merge tensors alias into it.
// ---------------------------------------------------------------------------
extern "C" void kernel_launch(void* const* d_in, const int* in_sizes, int n_in,
                              void* d_out, int out_size, void* d_ws,
                              size_t ws_size, hipStream_t stream) {
  (void)in_sizes; (void)n_in; (void)out_size; (void)ws_size;
  const float* x      = (const float*)d_in[0];
  const float* W_qkv  = (const float*)d_in[1];
  const float* b_qkv  = (const float*)d_in[2];
  const float* W_proj = (const float*)d_in[3];
  const float* b_proj = (const float*)d_in[4];
  float* out = (float*)d_out;

  char* w = (char*)d_ws;
  size_t off = 0;
  auto alloc = [&](size_t bytes) -> void* {
    void* p = w + off;
    off += (bytes + 255) & ~(size_t)255;
    return p;
  };

  const long nX  = (long)MTOK * CDIM;
  const long nWq = (long)C3 * CDIM;
  const long nWp = (long)CDIM * CDIM;
  const long nM  = (long)BATCH * NHEAD * KPAD * HD;  // merged elems (padded)

  bf16*  xb     = (bf16*)alloc(nX * 2);
  bf16*  wqkvb  = (bf16*)alloc(nWq * 2);
  bf16*  wprojb = (bf16*)alloc(nWp * 2);
  float* phi    = (float*)alloc((long)MTOK * NHEAD * 4);
  int*   keep   = (int*)alloc((long)BATCH * KKEEP * 4);
  int*   slot   = (int*)alloc((long)MTOK * 4);
  float* cnt    = (float*)alloc((long)BATCH * KKEEP * 4);
  float* qm     = (float*)alloc(3 * nM * 4);  // qm|km|vm contiguous
  float* km     = qm + nM;
  float* vm     = km + nM;
  char*  region = (char*)alloc((long)MTOK * C3 * 4);  // 454 MB scratch
  float* qkv    = (float*)region;                     // live until merge_accum
  bf16*  qmb    = (bf16*)region;                      // aliases (qkv dead)
  bf16*  kmb    = qmb + nM;
  bf16*  vmt    = kmb + nM;                           // V transposed
  bf16*  ymb    = vmt + nM;                           // [MRED, CDIM] bf16
  float* yproj  = (float*)(ymb + (long)MRED * CDIM);  // [MRED, CDIM] f32

  // 1) bf16 conversions
  cvt_f32_bf16<<<(unsigned)((nX + 255) / 256), 256, 0, stream>>>(x, xb, nX);
  cvt_f32_bf16<<<(unsigned)((nWq + 255) / 256), 256, 0, stream>>>(W_qkv, wqkvb, nWq);
  cvt_f32_bf16<<<(unsigned)((nWp + 255) / 256), 256, 0, stream>>>(W_proj, wprojb, nWp);

  // 2) QKV GEMM: [36928,1024] x [3072,1024]^T  (32x64 per wave)
  {
    int waves = (MTOK / 32) * (C3 / 64);
    gemm_bf16_f32<<<(waves + 7) / 8, 256, 0, stream>>>(xb, wqkvb, b_qkv, qkv,
                                                       MTOK, C3, CDIM);
  }

  // 3) phi, top-K, assignment
  phi_kernel<<<(MTOK + 255) / 256, 256, 0, stream>>>(qkv, phi);
  topk_kernel<<<BATCH, 512, 0, stream>>>(phi, keep);
  assign_kernel<<<BATCH, 256, 0, stream>>>(phi, keep, slot);

  // 4) segment-mean merge
  hipMemsetAsync(qm, 0, (size_t)(3 * nM) * 4, stream);
  hipMemsetAsync(cnt, 0, (size_t)BATCH * KKEEP * 4, stream);
  merge_accum<<<MTOK, 256, 0, stream>>>(qkv, slot, qm, km, vm, cnt);
  merge_norm<<<(unsigned)((nM + 255) / 256), 256, 0, stream>>>(qm, km, vm, cnt,
                                                               qmb, kmb, vmt);

  // 5) attention on reduced tokens (async K-panel staging, 160KB LDS)
  (void)hipFuncSetAttribute((const void*)attn_kernel,
                            hipFuncAttributeMaxDynamicSharedMemorySize,
                            LDS_ATTN_TOTAL);
  attn_kernel<<<BATCH * NHEAD * ATTN_QGRP, ATTN_WAVES * 32, LDS_ATTN_TOTAL,
                stream>>>(qmb, kmb, vmt, ymb);

  // 6) output projection: [18496,1024] x [1024,1024]^T
  {
    int waves = (MRED / 32) * (CDIM / 64);
    gemm_bf16_f32<<<(waves + 7) / 8, 256, 0, stream>>>(ymb, wprojb, b_proj,
                                                       yproj, MRED, CDIM, CDIM);
  }

  // 7) unmerge gather -> d_out
  {
    long total = (long)MTOK * (CDIM / 4);
    unmerge_kernel<<<(unsigned)((total + 255) / 256), 256, 0, stream>>>(
        yproj, slot, out);
  }
}
